// BiLSTM_3899830304992
// MI455X (gfx1250) — compile-verified, matrix operand order checked by reading, not run on hardware
//
#include <hip/hip_runtime.h>
#include <hip/hip_bf16.h>

// ---------------------------------------------------------------------------
// BiLSTM  T=512 B=32 I=H=512  -> out [B,T,2H] ++ hidd [B,2H] ++ cell [B,2H]
//
// Phase 1: fp32 -> bf16 conversion of x, Wih, Whh (matrix pipe wants bf16)
// Phase 2: xg GEMM  [16384 x 2048] per dir, K=512, bf16 WMMA, 32x64 wave tiles
// Phase 3: persistent recurrence: 64 WGs (32/dir); each WG keeps its entire
//          Whh slice register-resident (16 B-fragments = 128 VGPRs/lane) and
//          runs 512 steps with a per-direction global spin barrier.
// ---------------------------------------------------------------------------

typedef __attribute__((ext_vector_type(16))) __bf16 v16bf;
typedef __attribute__((ext_vector_type(8)))  float  v8f;

union FragBF { uint4 q[2]; v16bf v; };

__device__ __forceinline__ unsigned short f2bf(float f) {
  unsigned int u = __float_as_uint(f);
  unsigned int r = u + 0x7FFFu + ((u >> 16) & 1u);   // round-to-nearest-even
  return (unsigned short)(r >> 16);
}
__device__ __forceinline__ float bf2f(unsigned short h) {
  return __uint_as_float(((unsigned int)h) << 16);
}
__device__ __forceinline__ float sigf(float x) {
  return 1.0f / (1.0f + __expf(-x));
}

// ---------------------------------------------------------------- convert
__global__ void k_f32_to_bf16(const float* __restrict__ src,
                              unsigned short* __restrict__ dst, int n) {
  int i = blockIdx.x * blockDim.x + threadIdx.x;
  if (i < n) dst[i] = f2bf(src[i]);
}

// ------------------------------------------------------------------- init
__global__ void k_init(const float* __restrict__ h0f, const float* __restrict__ h0b,
                       unsigned short* __restrict__ hbuf, int* __restrict__ sync) {
  int i = blockIdx.x * blockDim.x + threadIdx.x;   // 64*256 = 16384
  if (i < 16384) {
    hbuf[i]             = f2bf(h0f[i]);            // dir0, buf0
    hbuf[2 * 16384 + i] = f2bf(h0b[i]);            // dir1, buf0
  }
  if (i < 64) sync[i] = 0;
}

// ------------------------------------------------------------ xg GEMM
// One wave = 32(M) x 64(N) tile, K=512: per k-step 2 A-frags + 4 B-frags,
// 8 WMMAs.  32768 waves total -> 4096 blocks of 8 waves.
__global__ void __launch_bounds__(256)
k_xg_gemm(const unsigned short* __restrict__ xbf,      // [16384][512] bf16
          const unsigned short* __restrict__ wih_bf,   // [2][2048][512] bf16
          const float* __restrict__ bih_f, const float* __restrict__ bhh_f,
          const float* __restrict__ bih_b, const float* __restrict__ bhh_b,
          unsigned short* __restrict__ xg)             // [2][16384][2048] bf16
{
  const int lane = threadIdx.x & 31;
  const int lsub = lane & 15;
  const int lh   = lane >> 4;

  int gw     = blockIdx.x * 8 + (threadIdx.x >> 5);  // 0..32767 waves
  int dir    = gw >> 14;                             // 16384 waves per dir
  int w      = gw & 16383;
  int nstrip = w & 31;                               // 2048/64
  int mtile  = w >> 5;                               // 16384/32 = 512
  const int mbase = mtile * 32;
  const int nbase = nstrip * 64;

  const unsigned short* W  = wih_bf + (size_t)dir * (2048 * 512);
  unsigned short*       XG = xg + (size_t)dir * ((size_t)16384 * 2048);
  const float* bih = dir ? bih_b : bih_f;
  const float* bhh = dir ? bhh_b : bhh_f;

  v8f acc[2][4];
#pragma unroll
  for (int mi = 0; mi < 2; ++mi)
#pragma unroll
    for (int j = 0; j < 4; ++j) { v8f z = {}; acc[mi][j] = z; }

  const unsigned short* ap0 = xbf + (size_t)(mbase + lsub) * 512;
  const unsigned short* ap1 = xbf + (size_t)(mbase + 16 + lsub) * 512;

  for (int kk = 0; kk < 16; ++kk) {
    const int koff = kk * 32 + lh * 8;
    FragBF a0, a1;
    a0.q[0] = *(const uint4*)(ap0 + koff);
    a0.q[1] = *(const uint4*)(ap0 + koff + 16);
    a1.q[0] = *(const uint4*)(ap1 + koff);
    a1.q[1] = *(const uint4*)(ap1 + koff + 16);
    __builtin_prefetch(ap0 + koff + 32, 0, 3);       // next K tile, keep near
    __builtin_prefetch(ap1 + koff + 32, 0, 3);
#pragma unroll
    for (int j = 0; j < 4; ++j) {
      const int ncol = nbase + j * 16 + lsub;
      const unsigned short* bp = W + (size_t)ncol * 512 + koff;
      FragBF b;
      b.q[0] = *(const uint4*)(bp);
      b.q[1] = *(const uint4*)(bp + 16);
      acc[0][j] = __builtin_amdgcn_wmma_f32_16x16x32_bf16(
          false, a0.v, false, b.v, (short)0, acc[0][j], false, false);
      acc[1][j] = __builtin_amdgcn_wmma_f32_16x16x32_bf16(
          false, a1.v, false, b.v, (short)0, acc[1][j], false, false);
    }
  }

#pragma unroll
  for (int j = 0; j < 4; ++j) {
    const int n = nbase + j * 16 + lsub;
    const float bias = bih[n] + bhh[n];
#pragma unroll
    for (int mi = 0; mi < 2; ++mi)
#pragma unroll
      for (int r = 0; r < 8; ++r) {
        const int row = mbase + mi * 16 + r + lh * 8;  // C/D layout
        XG[(size_t)row * 2048 + n] = f2bf(acc[mi][j][r] + bias);
      }
  }
}

// ------------------------------------------------------------ recurrence
// 64 persistent WGs: dir = blk>>5, slice = blk&31 (hidden units slice*16..+15).
// Whh slice is preloaded into registers (16 FragBF = 128 VGPRs/lane) and
// reused for all 512 steps; per step only h_prev (32KB, L2) is loaded.
#define NWG_PER_DIR 32
__global__ void __launch_bounds__(256)
k_recur(const unsigned short* __restrict__ whh_bf,  // [2][2048][512] bf16
        const unsigned short* __restrict__ xg,      // [2][16384][2048] bf16
        unsigned short* __restrict__ hbuf,          // [2][2][32*512] bf16
        int* __restrict__ sync,
        const float* __restrict__ c0f, const float* __restrict__ c0b,
        float* __restrict__ out)
{
  __shared__ float gates[32][64];   // [batch][gate*16 + jj]
  __shared__ float csh[32][16];     // cell state slice

  const int tid  = threadIdx.x;
  const int lane = tid & 31;
  const int lsub = lane & 15;
  const int lh   = lane >> 4;
  const int wave = tid >> 5;        // 0..7  (wave32)
  const int mt   = wave & 1;        // batch tile 0/1
  const int gt   = wave >> 1;       // gate 0..3 (i,f,g,o)

  const int dir   = blockIdx.x >> 5;
  const int slice = blockIdx.x & 31;

  const unsigned short* W  = whh_bf + (size_t)dir * (2048 * 512);
  const unsigned short* XG = xg + (size_t)dir * ((size_t)16384 * 2048);
  unsigned short* hb = hbuf + (size_t)dir * (2 * 16384);
  const float* c0 = dir ? c0b : c0f;
  int* sc = sync + dir;

  // cell init from c0
  for (int p = tid; p < 32 * 16; p += 256) {
    int b = p >> 4, jj = p & 15;
    csh[b][jj] = c0[b * 512 + slice * 16 + jj];
  }
  __syncthreads();

  // --- preload the full Whh slice for this wave into registers -------------
  // B row (fixed per wave): Whh row = gt*512 + slice*16 + lsub, all K=512.
  const unsigned short* wrow = W + (size_t)(gt * 512 + slice * 16 + lsub) * 512;
  FragBF bw[16];
#pragma unroll
  for (int kk = 0; kk < 16; ++kk) {
    const int koff = kk * 32 + lh * 8;
    bw[kk].q[0] = *(const uint4*)(wrow + koff);
    bw[kk].q[1] = *(const uint4*)(wrow + koff + 16);
  }

  const size_t OUT_HID  = (size_t)32 * 512 * 1024;  // after output [B,T,2H]
  const size_t OUT_CELL = OUT_HID + (size_t)32 * 1024;

  for (int s = 0; s < 512; ++s) {
    const int t = dir ? (511 - s) : s;
    const unsigned short* hp = hb + (s & 1) * 16384;       // read buffer
    unsigned short*       hn = hb + ((s + 1) & 1) * 16384; // write buffer

    // gates tile = h_prev[mt*16..+15, :] @ Whh_slice^T ; two accumulation
    // chains to shorten the serial WMMA dependence.
    v8f acc0 = {}, acc1 = {};
    const unsigned short* ar = hp + (size_t)(mt * 16 + lsub) * 512;
#pragma unroll
    for (int kk = 0; kk < 16; kk += 2) {
      const int koff0 = kk * 32 + lh * 8;
      const int koff1 = koff0 + 32;
      FragBF a0, a1;
      a0.q[0] = *(const uint4*)(ar + koff0);
      a0.q[1] = *(const uint4*)(ar + koff0 + 16);
      a1.q[0] = *(const uint4*)(ar + koff1);
      a1.q[1] = *(const uint4*)(ar + koff1 + 16);
      acc0 = __builtin_amdgcn_wmma_f32_16x16x32_bf16(
          false, a0.v, false, bw[kk].v,     (short)0, acc0, false, false);
      acc1 = __builtin_amdgcn_wmma_f32_16x16x32_bf16(
          false, a1.v, false, bw[kk + 1].v, (short)0, acc1, false, false);
    }
#pragma unroll
    for (int r = 0; r < 8; ++r)
      gates[mt * 16 + r + lh * 8][gt * 16 + lsub] = acc0[r] + acc1[r];
    __syncthreads();

    // elementwise LSTM update: 512 (b,jj) pairs over 256 threads
    for (int p = tid; p < 512; p += 256) {
      const int b = p >> 4, jj = p & 15;
      const unsigned short* xgr =
          XG + (size_t)(t * 32 + b) * 2048 + slice * 16 + jj;
      float gi = gates[b][ 0 + jj] + bf2f(xgr[0]);
      float gf = gates[b][16 + jj] + bf2f(xgr[512]);
      float gg = gates[b][32 + jj] + bf2f(xgr[1024]);
      float go = gates[b][48 + jj] + bf2f(xgr[1536]);
      float c = sigf(gf) * csh[b][jj] + sigf(gi) * tanhf(gg);
      float h = sigf(go) * tanhf(c);
      csh[b][jj] = c;
      const int hid = slice * 16 + jj;
      hn[b * 512 + hid] = f2bf(h);
      out[(size_t)b * (512 * 1024) + (size_t)t * 1024 + dir * 512 + hid] = h;
      if (s == 511) {                                  // final states
        out[OUT_HID  + (size_t)b * 1024 + dir * 512 + hid] = h;
        out[OUT_CELL + (size_t)b * 1024 + dir * 512 + hid] = c;
      }
    }
    __threadfence();
    __syncthreads();
    if (tid == 0) {                                    // grid barrier (per dir)
      atomicAdd(sc, 1);
      const int target = (s + 1) * NWG_PER_DIR;        // monotonic counter
      while (atomicAdd(sc, 0) < target) { __builtin_amdgcn_s_sleep(1); }
    }
    __syncthreads();
  }
}

// ---------------------------------------------------------------------------
extern "C" void kernel_launch(void* const* d_in, const int* in_sizes, int n_in,
                              void* d_out, int out_size, void* d_ws, size_t ws_size,
                              hipStream_t stream) {
  (void)in_sizes; (void)n_in; (void)out_size; (void)ws_size;

  const float* inp   = (const float*)d_in[0];
  const float* Wih_f = (const float*)d_in[1];
  const float* Whh_f = (const float*)d_in[2];
  const float* bih_f = (const float*)d_in[3];
  const float* bhh_f = (const float*)d_in[4];
  const float* Wih_b = (const float*)d_in[5];
  const float* Whh_b = (const float*)d_in[6];
  const float* bih_b = (const float*)d_in[7];
  const float* bhh_b = (const float*)d_in[8];
  const float* h0f   = (const float*)d_in[9];
  const float* c0f   = (const float*)d_in[10];
  const float* h0b   = (const float*)d_in[11];
  const float* c0b   = (const float*)d_in[12];
  float* out = (float*)d_out;

  char* ws = (char*)d_ws;
  size_t o = 0;
  unsigned short* xbf    = (unsigned short*)(ws + o); o += (size_t)16384 * 512 * 2;       // 16 MB
  unsigned short* wih_bf = (unsigned short*)(ws + o); o += (size_t)2 * 2048 * 512 * 2;    //  4 MB
  unsigned short* whh_bf = (unsigned short*)(ws + o); o += (size_t)2 * 2048 * 512 * 2;    //  4 MB
  unsigned short* xg     = (unsigned short*)(ws + o); o += (size_t)2 * 16384 * 2048 * 2;  // 128 MB
  unsigned short* hbuf   = (unsigned short*)(ws + o); o += (size_t)2 * 2 * 16384 * 2;     // 128 KB
  int*            syncc  = (int*)(ws + o);

  // Phase 1: fp32 -> bf16
  k_f32_to_bf16<<<(8388608 + 255) / 256, 256, 0, stream>>>(inp, xbf, 8388608);
  k_f32_to_bf16<<<4096, 256, 0, stream>>>(Wih_f, wih_bf,            1048576);
  k_f32_to_bf16<<<4096, 256, 0, stream>>>(Wih_b, wih_bf + 1048576,  1048576);
  k_f32_to_bf16<<<4096, 256, 0, stream>>>(Whh_f, whh_bf,            1048576);
  k_f32_to_bf16<<<4096, 256, 0, stream>>>(Whh_b, whh_bf + 1048576,  1048576);

  // init h buffers + barrier counters (reset every call: graph-replay safe)
  k_init<<<64, 256, 0, stream>>>(h0f, h0b, hbuf, syncc);

  // Phase 2: input-projection GEMM (both directions, 32x64 wave tiles)
  k_xg_gemm<<<4096, 256, 0, stream>>>(xbf, wih_bf, bih_f, bhh_f, bih_b, bhh_b, xg);

  // Phase 3: persistent recurrence (32 WGs per direction)
  k_recur<<<64, 256, 0, stream>>>(whh_bf, xg, hbuf, syncc, c0f, c0b, out);
}
